// VTIRTMultiKC_20658792694284
// MI455X (gfx1250) — matrix operand order repaired and checked by reading, not compiled, not run on hardware
//
#include <hip/hip_runtime.h>
#include <hip/hip_bf16.h>
#include <stdint.h>

// Problem constants (from reference)
#define U_DIM 2048
#define T_DIM 200
#define Q_DIM 4096
#define K_DIM 16
#define H_DIM 256
#define N_ROWS (U_DIM * T_DIM)   // 409600
#define LT 1.0f                   // LMDA_THETA

typedef __attribute__((ext_vector_type(16))) _Float16 v16h;
typedef __attribute__((ext_vector_type(8)))  float    v8f;

union Frag16 {
    v16h   v;
    float4 f4[2];
};

__device__ __forceinline__ float gelu_exact(float x) {
    // jax.nn.gelu(approximate=False): 0.5*x*(1+erf(x/sqrt(2)))
    return 0.5f * x * (1.0f + erff(x * 0.70710678118654752440f));
}

// ---------------------------------------------------------------------------
// Kernel 0: W2 (H x H, f32, row-major, k-major) -> W2T (f16, column-major):
//   W2T[c*H + k] = (f16) W2[k*H + c]
// This makes each lane's WMMA B-fragment two contiguous 16B runs along k.
// ---------------------------------------------------------------------------
__global__ void prep_w2t_kernel(const float* __restrict__ W2,
                                _Float16* __restrict__ W2T) {
    int c = blockIdx.x;      // 0..255  output column of layer-2
    int k = threadIdx.x;     // 0..255  reduction index
    W2T[c * H_DIM + k] = (_Float16)W2[k * H_DIM + c];
}

// ---------------------------------------------------------------------------
// Kernel 1: fused MLP. Block = 256 threads (8 waves), handles 64 rows.
//   h1 = gelu(inp @ W1 + b1)          (into LDS, f16)
//   h2 = gelu(h1  @ W2 + b2)          (WMMA f16 -> f32 acc, back into LDS)
//   poten = gelu(h2 @ W3 + b3)        (tiny, per-row VALU)
//   mu = poten[:,0]; lmda = exp(-max(poten[:,1], 1e-8))
// ---------------------------------------------------------------------------
#define MBLK 64

__global__ void
__launch_bounds__(256, 2)
mlp_kernel(const int*   __restrict__ q_id,     // U*T
           const float* __restrict__ resp,     // U*T
           const float* __restrict__ diff_mu,  // Q
           const float* __restrict__ disc_mu,  // Q
           const float* __restrict__ W1,       // 3 x H
           const float* __restrict__ b1,       // H
           const _Float16* __restrict__ W2T,   // H x H, column-major
           const float* __restrict__ b2,       // H
           const float* __restrict__ W3,       // H x 2
           const float* __restrict__ b3,       // 2
           float* __restrict__ mu_ws,          // U*T
           float* __restrict__ lmda_ws)        // U*T
{
    __shared__ _Float16 sh[MBLK * H_DIM];      // 32 KB, reused h1 -> h2
    __shared__ float    sfeat[MBLK * 3];       // td, ts, resp per row

    const int tid     = threadIdx.x;
    const int rowBase = blockIdx.x * MBLK;

    // ---- Phase 0: gather per-row features -------------------------------
    if (tid < MBLK) {
        int n = rowBase + tid;
        int q = q_id[n];
        sfeat[tid * 3 + 0] = diff_mu[q];
        sfeat[tid * 3 + 1] = disc_mu[q];
        sfeat[tid * 3 + 2] = resp[n];
    }
    __syncthreads();

    // ---- Phase 1: h1 = gelu(inp @ W1 + b1) into LDS (f16) ---------------
    {
        const int c   = tid;                   // each thread owns one column
        const float w0 = W1[0 * H_DIM + c];
        const float w1 = W1[1 * H_DIM + c];
        const float w2 = W1[2 * H_DIM + c];
        const float bb = b1[c];
#pragma unroll 4
        for (int r = 0; r < MBLK; ++r) {
            float x = sfeat[r * 3 + 0] * w0 +
                      sfeat[r * 3 + 1] * w1 +
                      sfeat[r * 3 + 2] * w2 + bb;
            sh[r * H_DIM + c] = (_Float16)gelu_exact(x);
        }
    }
    __syncthreads();

    // ---- Phase 2: h2 = h1 @ W2 via v_wmma_f32_16x16x32_f16 --------------
    // Wave w: row-tile rt = w&3 (16 rows), column half cHalf = w>>2 (128 cols
    // = 8 column tiles). 8 k-steps of 32.
    const int wv    = tid >> 5;
    const int lane  = tid & 31;
    const int rt    = wv & 3;
    const int cHalf = wv >> 2;
    const int laneR = lane & 15;
    const int hi    = lane >> 4;               // 0: lanes 0-15, 1: lanes 16-31

    v8f acc[8];
#pragma unroll
    for (int ct = 0; ct < 8; ++ct)
#pragma unroll
        for (int j = 0; j < 8; ++j) acc[ct][j] = 0.0f;

    const _Float16* aRow = &sh[(rt * 16 + laneR) * H_DIM];

#pragma unroll
    for (int ks = 0; ks < 8; ++ks) {
        const int kBase = ks * 32;
        // A fragment (16x32 f16): lane<16 -> K 0..7 & 16..23 of kBase,
        //                         lane>=16 -> K 8..15 & 24..31.
        Frag16 a;
        a.f4[0] = *(const float4*)(aRow + kBase + hi * 8);
        a.f4[1] = *(const float4*)(aRow + kBase + 16 + hi * 8);

#pragma unroll
        for (int ct = 0; ct < 8; ++ct) {
            const int col = cHalf * 128 + ct * 16 + laneR;
            // B fragment (32x16 f16): lane<16 -> K 0..7 & 16..23, lane>=16 -> K 8..15 & 24..31
            const _Float16* bPtr = W2T + col * H_DIM + kBase + hi * 8;
            Frag16 b;
            b.f4[0] = *(const float4*)(bPtr);
            b.f4[1] = *(const float4*)(bPtr + 16);
            acc[ct] = __builtin_amdgcn_wmma_f32_16x16x32_f16(
                false, a.v, false, b.v, (short)0, acc[ct], false, false);
        }
    }
    __syncthreads();   // all A reads done before overwriting sh with h2

    // ---- Phase 2.5: bias + gelu, h2 back into LDS (f16) -----------------
#pragma unroll
    for (int ct = 0; ct < 8; ++ct) {
        const int col  = cHalf * 128 + ct * 16 + laneR;
        const float bb = b2[col];
#pragma unroll
        for (int j = 0; j < 8; ++j) {
            // D layout: VGPR j, lanes 0-15 -> M=j ; lanes 16-31 -> M=8+j
            const int row = rt * 16 + j + hi * 8;
            sh[row * H_DIM + col] = (_Float16)gelu_exact(acc[ct][j] + bb);
        }
    }
    __syncthreads();

    // ---- Phase 3: layer 3 (256 -> 2), mu / lmda -------------------------
    if (tid < MBLK) {
        const _Float16* hrow = &sh[tid * H_DIM];
        float d0 = b3[0];
        float d1 = b3[1];
#pragma unroll 4
        for (int c = 0; c < H_DIM; ++c) {
            float hv = (float)hrow[c];
            d0 += hv * W3[c * 2 + 0];
            d1 += hv * W3[c * 2 + 1];
        }
        const float p0 = gelu_exact(d0);
        const float p1 = gelu_exact(d1);
        const int n = rowBase + tid;
        mu_ws[n]   = p0;
        lmda_ws[n] = expf(-fmaxf(p1, 1e-8f));
    }
}

// ---------------------------------------------------------------------------
// Kernel 2: backward + forward scans. One thread per (u,k) chain.
// 16 consecutive lanes share a user u -> K-reduction via __shfl_xor in-wave.
// alphas stored [t][u*16+k] (coalesced, L2-resident: ~26 MB).
// ---------------------------------------------------------------------------
__global__ void
__launch_bounds__(256)
scan_kernel(const int*     __restrict__ q_id,    // U*T
            const uint8_t* __restrict__ kmap,    // Q*K (bool)
            const float*   __restrict__ mu_ws,   // U*T
            const float*   __restrict__ lmda_ws, // U*T
            float*         __restrict__ alphas,  // T * (U*K)
            float*         __restrict__ out)     // U*T
{
    const int g = blockIdx.x * blockDim.x + threadIdx.x;  // 0 .. U*K-1
    const int u = g >> 4;
    const int k = g & 15;
    const int UK = U_DIM * K_DIM;

    // ---- backward scan (t = T-1 .. 0), store alphas ---------------------
    float alpha = 0.0f, beta = 0.0f;
    for (int t = T_DIM - 1; t >= 0; --t) {
        const int n = u * T_DIM + t;
        const float l   = lmda_ws[n];
        const float m_  = mu_ws[n];
        const bool  msk = kmap[q_id[n] * K_DIM + k] != 0;
        const float an = (l + alpha * LT) / (LT + l + alpha * LT);
        const float bn = (l * m_ + an * l * beta) / (l + an * l);
        if (msk) { alpha = an; beta = bn; }
        alphas[t * UK + g] = alpha;
    }

    // ---- forward scan (t = 0 .. T-1) + masked mean over K ---------------
    float ability = 0.0f;
    for (int t = 0; t < T_DIM; ++t) {
        const int n = u * T_DIM + t;
        const float l   = lmda_ws[n];
        const float m_  = mu_ws[n];
        const bool  msk = kmap[q_id[n] * K_DIM + k] != 0;
        const float a_nxt = (t + 1 < T_DIM) ? alphas[(t + 1) * UK + g] : 0.0f;
        // b_nxt == a_nxt in the reference
        const float mu_t = (LT * ability + m_ * l + a_nxt * LT * a_nxt) /
                           (LT + l + a_nxt * LT);
        const float ab = msk ? mu_t : ability;
        ability = ab;

        const float mf = msk ? 1.0f : 0.0f;
        float num = ab * mf;
        float den = mf;
        // reduce over the 16 lanes sharing user u (stays inside half-wave)
#pragma unroll
        for (int o = 1; o < 16; o <<= 1) {
            num += __shfl_xor(num, o, 32);
            den += __shfl_xor(den, o, 32);
        }
        if (k == 0) out[n] = num / fmaxf(den, 1e-8f);
    }
}

// ---------------------------------------------------------------------------
// Launch
// ---------------------------------------------------------------------------
extern "C" void kernel_launch(void* const* d_in, const int* in_sizes, int n_in,
                              void* d_out, int out_size, void* d_ws, size_t ws_size,
                              hipStream_t stream) {
    // setup_inputs order:
    // 0: mask (bool, unused)  1: q_id (int)   2: kmap (bool)   3: resp (f32)
    // 4: diff_mu (f32)        5: disc_mu      6: W1            7: b1
    // 8: W2                   9: b2          10: W3           11: b3
    const int*     q_id    = (const int*)d_in[1];
    const uint8_t* kmap    = (const uint8_t*)d_in[2];
    const float*   resp    = (const float*)d_in[3];
    const float*   diff_mu = (const float*)d_in[4];
    const float*   disc_mu = (const float*)d_in[5];
    const float*   W1      = (const float*)d_in[6];
    const float*   b1      = (const float*)d_in[7];
    const float*   W2      = (const float*)d_in[8];
    const float*   b2      = (const float*)d_in[9];
    const float*   W3      = (const float*)d_in[10];
    const float*   b3      = (const float*)d_in[11];
    float*         out     = (float*)d_out;

    // workspace layout
    char* ws = (char*)d_ws;
    _Float16* W2T     = (_Float16*)ws;                         // 256*256*2 = 128 KB
    float*    mu_ws   = (float*)(ws + H_DIM * H_DIM * sizeof(_Float16));
    float*    lmda_ws = mu_ws + N_ROWS;
    float*    alphas  = lmda_ws + N_ROWS;                      // T * U * K floats

    // k0: W2 -> W2T (f16, column-major)
    prep_w2t_kernel<<<H_DIM, H_DIM, 0, stream>>>(W2, W2T);

    // k1: fused MLP with WMMA
    mlp_kernel<<<N_ROWS / MBLK, 256, 0, stream>>>(
        q_id, resp, diff_mu, disc_mu, W1, b1, W2T, b2, W3, b3, mu_ws, lmda_ws);

    // k2: backward + forward scans -> output
    scan_kernel<<<(U_DIM * K_DIM) / 256, 256, 0, stream>>>(
        q_id, kmap, mu_ws, lmda_ws, alphas, out);
}
